// LigerLMHeadCE_64553358459089
// MI455X (gfx1250) — matrix operand order repaired and checked by reading, not compiled, not run on hardware
//
#include <hip/hip_runtime.h>
#include <hip/hip_bf16.h>
#include <math.h>

// ---------------------------------------------------------------------------
// Fused LM-head cross-entropy for MI455X (gfx1250, wave32, WMMA).
//   loss = mean_over_valid( logsumexp(x @ W^T, axis=V) - logit[y] )
// Compute-bound (AI ~ 2048 FLOP/B) -> v_wmma_f32_16x16x32_bf16 with f32 acc.
// ---------------------------------------------------------------------------

typedef __attribute__((ext_vector_type(16))) __bf16 v16bf;
typedef __attribute__((ext_vector_type(8)))  float  v8f;

#define HDIM        4096
#define NROWS       4096
#define VOCAB       128000
#define ROWS_PER_WG 16
#define WAVES       8
#define THREADS     (WAVES * 32)
#define COLS_PER_IT (WAVES * 16)      // 128 -> 1000 outer iterations
#define XPAD        8                 // bf16 elements of row padding (bank skew)
#define XSTRIDE     (HDIM + XPAD)     // 4104
#define IGNORE_IDX  (-100)

union V16U { v16bf v; uint4 q[2]; };

static __device__ __forceinline__ unsigned short f32_to_bf16_rne(float f) {
  unsigned u = __float_as_uint(f);
  unsigned r = u + 0x7FFFu + ((u >> 16) & 1u);   // round-to-nearest-even
  return (unsigned short)(r >> 16);
}
static __device__ __forceinline__ float bf16_to_f32(unsigned short h) {
  return __uint_as_float(((unsigned)h) << 16);
}

// ---------------------------------------------------------------------------
// Kernel 1: W fp32 -> bf16 into workspace (amortizes conversion, halves the
// streamed footprint so the 256-WG cooperative V-sweep fits the 192MB L2
// window). Also zeroes the (loss_sum, valid_count) accumulators.
// ---------------------------------------------------------------------------
__global__ void LigerCE_convertW(const float* __restrict__ W,
                                 unsigned short* __restrict__ Wb,
                                 float* __restrict__ accum, size_t n4) {
  size_t i = (size_t)blockIdx.x * blockDim.x + threadIdx.x;
  if (i == 0) { accum[0] = 0.0f; accum[1] = 0.0f; }
  size_t stride = (size_t)gridDim.x * blockDim.x;
  const float4* W4 = (const float4*)W;
  ushort2* Wb2 = (ushort2*)Wb;
  for (size_t j = i; j < n4; j += stride) {
    float4 f = W4[j];
    ushort2 a; a.x = f32_to_bf16_rne(f.x); a.y = f32_to_bf16_rne(f.y);
    ushort2 b; b.x = f32_to_bf16_rne(f.z); b.y = f32_to_bf16_rne(f.w);
    Wb2[2 * j]     = a;
    Wb2[2 * j + 1] = b;
  }
}

// ---------------------------------------------------------------------------
// Kernel 2: target logits. tgt[r] = dot(bf16(x[r]), Wb[y[r]]) in f32.
// One wave32 per row; <0.001% of total FLOPs.
// ---------------------------------------------------------------------------
__global__ void LigerCE_target(const float* __restrict__ x,
                               const int* __restrict__ y,
                               const unsigned short* __restrict__ Wb,
                               float* __restrict__ tgtg) {
  int row  = blockIdx.x * (blockDim.x >> 5) + (threadIdx.x >> 5);
  int lane = threadIdx.x & 31;
  if (row >= NROWS) return;
  int yy = y[row];
  float acc = 0.0f;
  if (yy != IGNORE_IDX) {
    const float* xr = x + (size_t)row * HDIM;
    const unsigned short* wr = Wb + (size_t)yy * HDIM;
    for (int k = lane * 4; k < HDIM; k += 32 * 4) {
      float4  xv = *(const float4*)(xr + k);
      ushort4 wv = *(const ushort4*)(wr + k);
      acc += bf16_to_f32(f32_to_bf16_rne(xv.x)) * bf16_to_f32(wv.x);
      acc += bf16_to_f32(f32_to_bf16_rne(xv.y)) * bf16_to_f32(wv.y);
      acc += bf16_to_f32(f32_to_bf16_rne(xv.z)) * bf16_to_f32(wv.z);
      acc += bf16_to_f32(f32_to_bf16_rne(xv.w)) * bf16_to_f32(wv.w);
    }
#pragma unroll
    for (int off = 16; off; off >>= 1) acc += __shfl_xor(acc, off, 32);
  }
  if (lane == 0) tgtg[row] = acc;
}

// ---------------------------------------------------------------------------
// Kernel 3: main fused GEMM + online logsumexp.
// Each WG: 16 rows of x (bf16 tile resident in LDS), sweeps all V columns.
// Per wave: one 16-col slice per outer iteration; depth-2 pipelined K-loop.
// ---------------------------------------------------------------------------
__launch_bounds__(THREADS)
__global__ void LigerCE_main(const float* __restrict__ x,
                             const int* __restrict__ y,
                             const unsigned short* __restrict__ Wb,
                             const float* __restrict__ tgtg,
                             float* __restrict__ accum) {
  extern __shared__ char smem_raw[];
  unsigned short* xs = (unsigned short*)smem_raw;                      // [16][XSTRIDE] bf16
  float* wred        = (float*)(smem_raw + ROWS_PER_WG * XSTRIDE * 2); // [WAVES][16][2]

  const int tid  = threadIdx.x;
  const int wave = tid >> 5;
  const int lane = tid & 31;
  const int lo   = lane & 15;    // B column within wave-tile / A row
  const int hi   = lane >> 4;    // half-wave selector
  const int r0   = blockIdx.x * ROWS_PER_WG;

  // ---- phase 0: stage x tile (fp32 -> bf16) into LDS ----
  for (int idx = tid; idx < ROWS_PER_WG * (HDIM / 4); idx += THREADS) {
    int row = idx / (HDIM / 4);
    int c4  = (idx % (HDIM / 4)) * 4;
    float4 f = *(const float4*)(x + (size_t)(r0 + row) * HDIM + c4);
    ushort4 pk;
    pk.x = f32_to_bf16_rne(f.x); pk.y = f32_to_bf16_rne(f.y);
    pk.z = f32_to_bf16_rne(f.z); pk.w = f32_to_bf16_rne(f.w);
    *(ushort4*)(xs + row * XSTRIDE + c4) = pk;
  }
  __syncthreads();

  // online-softmax state: lane holds rows v+8*hi (C layout), one column subset
  float m[8], s[8];
#pragma unroll
  for (int v = 0; v < 8; v++) { m[v] = -INFINITY; s[v] = 0.0f; }

#define LOAD_AB(A_, B_, K_) do {                                  \
    (A_).q[0] = *(const uint4*)(ap_base + (K_));                  \
    (A_).q[1] = *(const uint4*)(ap_base + (K_) + 16);             \
    (B_).q[0] = *(const uint4*)(bp + (K_));                       \
    (B_).q[1] = *(const uint4*)(bp + (K_) + 16);                  \
  } while (0)

  // ---- sweep all vocab columns ----
  for (int vb = 0; vb < VOCAB; vb += COLS_PER_IT) {
    // Keep the 8 waves converged on the same 1MB W window (L2-cooperative
    // sweep) and fence LDS so the x-tile loads cannot be hoisted out of
    // this loop (LICM previously hoisted+spilled the whole A tile).
    __syncthreads();

    // Opaque LDS base offset: defeats loop-invariant hoisting of A loads.
    // A layout (16-bit): lanes 0-15 K={0..7,16..23}, lanes 16-31 K={8..15,24..31}.
    unsigned abase = (unsigned)(lo * XSTRIDE + hi * 8);
    asm volatile("" : "+v"(abase));
    const unsigned short* ap_base = xs + abase;

    const int colg = vb + wave * 16 + lo;
    // B layout (32x16 bf16): lane = column colg; lanes 0-15 K={0..7,16..23},
    // lanes 16-31 K={8..15,24..31} (matches A).
    const unsigned short* bp = Wb + (size_t)colg * HDIM + hi * 8;

    V16U a0, b0, a1, b1;
    LOAD_AB(a0, b0, 0);
    LOAD_AB(a1, b1, 32);
    v8f c = {};

    // depth-2 software pipeline: WMMA on buffer i while buffer i+2 loads
#pragma unroll 2
    for (int k = 0; k < HDIM - 64; k += 64) {
      c = __builtin_amdgcn_wmma_f32_16x16x32_bf16(
              false, a0.v, false, b0.v, (short)0, c, false, false);
      LOAD_AB(a0, b0, k + 64);
      // prefetch the NEXT column tile's B row: 128B/step x 64 steps = 8KB row
      __builtin_prefetch(bp + (size_t)COLS_PER_IT * HDIM + k, 0, 3);
      c = __builtin_amdgcn_wmma_f32_16x16x32_bf16(
              false, a1.v, false, b1.v, (short)0, c, false, false);
      LOAD_AB(a1, b1, k + 96);
    }
    c = __builtin_amdgcn_wmma_f32_16x16x32_bf16(
            false, a0.v, false, b0.v, (short)0, c, false, false);
    c = __builtin_amdgcn_wmma_f32_16x16x32_bf16(
            false, a1.v, false, b1.v, (short)0, c, false, false);

    // online logsumexp update (cheap vs 128 WMMAs)
#pragma unroll
    for (int v = 0; v < 8; v++) {
      float l  = c[v];
      float nm = fmaxf(m[v], l);
      s[v] = s[v] * __expf(m[v] - nm) + __expf(l - nm);
      m[v] = nm;
    }
  }
#undef LOAD_AB

  // ---- reduce (m,s) across the 16 lanes of each half-wave (column groups) ----
#pragma unroll
  for (int off = 1; off < 16; off <<= 1) {
#pragma unroll
    for (int v = 0; v < 8; v++) {
      float om = __shfl_xor(m[v], off, 32);
      float os = __shfl_xor(s[v], off, 32);
      float nm = fmaxf(m[v], om);
      s[v] = s[v] * __expf(m[v] - nm) + os * __expf(om - nm);
      m[v] = nm;
    }
  }
  if (lo == 0) {               // lanes 0 (rows 0-7) and 16 (rows 8-15)
#pragma unroll
    for (int v = 0; v < 8; v++) {
      int row = v + 8 * hi;
      wred[(wave * ROWS_PER_WG + row) * 2 + 0] = m[v];
      wred[(wave * ROWS_PER_WG + row) * 2 + 1] = s[v];
    }
  }
  __syncthreads();

  // ---- merge across waves, finish per-row loss, accumulate globally ----
  if (tid < ROWS_PER_WG) {
    float mm = -INFINITY, ss = 0.0f;
#pragma unroll
    for (int w = 0; w < WAVES; w++) {
      float om = wred[(w * ROWS_PER_WG + tid) * 2 + 0];
      float os = wred[(w * ROWS_PER_WG + tid) * 2 + 1];
      float nm = fmaxf(mm, om);
      ss = ss * __expf(mm - nm) + os * __expf(om - nm);
      mm = nm;
    }
    float lse = mm + __logf(ss);
    if (y[r0 + tid] != IGNORE_IDX) {
      atomicAdd(&accum[0], lse - tgtg[r0 + tid]);  // global_atomic_add_f32
      atomicAdd(&accum[1], 1.0f);
    }
  }
}

// ---------------------------------------------------------------------------
// Kernel 4: mean reduction -> d_out[0]
// ---------------------------------------------------------------------------
__global__ void LigerCE_finalize(const float* __restrict__ accum,
                                 float* __restrict__ out) {
  out[0] = accum[0] / fmaxf(accum[1], 1.0f);
}

// ---------------------------------------------------------------------------
extern "C" void kernel_launch(void* const* d_in, const int* in_sizes, int n_in,
                              void* d_out, int out_size, void* d_ws, size_t ws_size,
                              hipStream_t stream) {
  (void)in_sizes; (void)n_in; (void)out_size; (void)ws_size;

  const float* x = (const float*)d_in[0];   // [N,H] fp32
  const int*   y = (const int*)d_in[1];     // [N] labels
  const float* W = (const float*)d_in[2];   // [V,H] fp32

  unsigned short* Wb = (unsigned short*)d_ws;                        // [V,H] bf16
  char* tail  = (char*)d_ws + (size_t)VOCAB * HDIM * 2;
  float* accum = (float*)tail;                                       // {sum, count}
  float* tgtg  = (float*)(tail + 16);                                // [N] target logits

  // Raise dynamic LDS limit for the 131 KB x-tile (WGP allows up to 320 KB).
  size_t lds_bytes = (size_t)ROWS_PER_WG * XSTRIDE * 2
                   + WAVES * ROWS_PER_WG * 2 * sizeof(float);        // wred
  (void)hipFuncSetAttribute((const void*)LigerCE_main,
                            hipFuncAttributeMaxDynamicSharedMemorySize,
                            (int)lds_bytes);

  // 1) convert W to bf16 once + zero accumulators
  size_t n4 = (size_t)VOCAB * HDIM / 4;
  LigerCE_convertW<<<4096, 256, 0, stream>>>(W, Wb, accum, n4);

  // 2) per-row target logits (one wave per row)
  LigerCE_target<<<NROWS / (THREADS / 32), THREADS, 0, stream>>>(x, y, Wb, tgtg);

  // 3) fused GEMM + online logsumexp (256 WGs x 8 wave32)
  LigerCE_main<<<NROWS / ROWS_PER_WG, THREADS, lds_bytes, stream>>>(x, y, Wb, tgtg, accum);

  // 4) mean
  LigerCE_finalize<<<1, 1, 0, stream>>>(accum, (float*)d_out);
}
